// CausalSelfAttention_39737037423347
// MI455X (gfx1250) — matrix operand order repaired and checked by reading, not compiled, not run on hardware
//
#include <hip/hip_runtime.h>
#include <hip/hip_bf16.h>

typedef __bf16 bf16;
typedef __attribute__((ext_vector_type(8)))  __bf16 v8bf;
typedef __attribute__((ext_vector_type(16))) __bf16 v16bf;
typedef __attribute__((ext_vector_type(8)))  float  v8f;

#define BB 4
#define TT 2048
#define CC 2048
#define NH 16
#define HD 128
constexpr int ROWS = BB * TT;   // 8192 tokens
constexpr int N3C  = 3 * CC;    // 6144

__device__ __forceinline__ v8f wmma_bf16(v16bf a, v16bf b, v8f c) {
    // (neg_a, A, neg_b, B, c_mod, C, reuse_a, reuse_b)
    return __builtin_amdgcn_wmma_f32_16x16x32_bf16(false, a, false, b, (short)0, c,
                                                   false, false);
}
__device__ __forceinline__ v8f zero8() {
    v8f z = {0.f, 0.f, 0.f, 0.f, 0.f, 0.f, 0.f, 0.f};
    return z;
}
__device__ __forceinline__ v16bf cat16(v8bf a, v8bf b) {
    return __builtin_shufflevector(a, b, 0, 1, 2, 3, 4, 5, 6, 7,
                                         8, 9, 10, 11, 12, 13, 14, 15);
}
__device__ __forceinline__ v16bf loadA(const bf16* p) {
    return cat16(*(const v8bf*)p, *(const v8bf*)(p + 8));
}

// ---------------------------------------------------------------- pack kernels
__global__ void cvt_bf16_kernel(const float* __restrict__ s, bf16* __restrict__ d, int n) {
    int i = blockIdx.x * blockDim.x + threadIdx.x;
    int stride = gridDim.x * blockDim.x;
    for (; i < n; i += stride) d[i] = (bf16)s[i];
}

// W (K x N, row major f32) -> Wt (N x K, row major bf16)
__global__ void transpose_bf16_kernel(const float* __restrict__ w, bf16* __restrict__ wt,
                                      int K, int N) {
    long i = (long)blockIdx.x * blockDim.x + threadIdx.x;
    long total = (long)K * N;
    long stride = (long)gridDim.x * blockDim.x;
    for (; i < total; i += stride) {
        long kk = i / N, n = i % N;
        wt[n * (long)K + kk] = (bf16)w[i];
    }
}

// ---------------------------------------------------------------- QKV GEMM + RoPE
// A: (8192 x 2048) bf16, Bt: (6144 x 2048) bf16 (N-major). One wave -> 16x64 tile.
// Depth-1 software pipeline: chunk c+32 loads issue while chunk c WMMAs execute.
__global__ void gemm_qkv_rope_kernel(const bf16* __restrict__ A, const bf16* __restrict__ Bt,
                                     const float* __restrict__ bias,
                                     bf16* __restrict__ qo, bf16* __restrict__ ko,
                                     bf16* __restrict__ vo) {
    const int lane = threadIdx.x & 31;
    const int hi = lane >> 4, ln = lane & 15;
    const int wid = blockIdx.x * (blockDim.x >> 5) + (threadIdx.x >> 5);
    const int mt = wid / 96, ng = wid % 96;          // 512 x 96 wave tiles
    const int m0 = mt * 16, n0 = ng * 64;

    v8f acc[4];
#pragma unroll
    for (int nt = 0; nt < 4; ++nt) acc[nt] = zero8();

    const bf16* ap = A + (size_t)(m0 + ln) * CC + hi * 16;   // A frag base (+8 for 2nd run)
    const bf16* bp = Bt + (size_t)(n0 + ln) * CC + hi * 16;  // nt rows: +nt*16*CC (64KB)

    v16bf af = loadA(ap);
    v16bf bfr[4];
#pragma unroll
    for (int nt = 0; nt < 4; ++nt) bfr[nt] = *(const v16bf*)(bp + nt * 16 * CC);
    ap += 32;
    bp += 32;

    for (int c = 0; c < CC - 32; c += 32) {
        v16bf af_n = loadA(ap);
        v16bf bfn[4];
#pragma unroll
        for (int nt = 0; nt < 4; ++nt) bfn[nt] = *(const v16bf*)(bp + nt * 16 * CC);
        ap += 32;
        bp += 32;
#pragma unroll
        for (int nt = 0; nt < 4; ++nt) acc[nt] = wmma_bf16(af, bfr[nt], acc[nt]);
        af = af_n;
#pragma unroll
        for (int nt = 0; nt < 4; ++nt) bfr[nt] = bfn[nt];
    }
#pragma unroll
    for (int nt = 0; nt < 4; ++nt) acc[nt] = wmma_bf16(af, bfr[nt], acc[nt]);

    // epilogue: bias + RoPE (q,k) + scatter
#pragma unroll
    for (int nt = 0; nt < 4; ++nt) {
        const int ncol = n0 + nt * 16 + ln;
        const int part = ncol / CC;                  // 0=q 1=k 2=v (64-tile never crosses)
        const int rem = ncol % CC;
        const int h = rem / HD, d = rem % HD;
        const float bv = bias[ncol];
        // inv_freq[d & 63] = 10000^(-(d&63)/64)
        const float invf = __expf(-9.210340371976184f * (float)(d & 63) * (1.0f / 64.0f));
#pragma unroll
        for (int r = 0; r < 8; ++r) {
            const int row = m0 + hi * 8 + r;         // C-frag: M = r + 8*hi
            const int b = row >> 11, t = row & (TT - 1);
            float val = acc[nt][r] + bv;
            float partner = __shfl_xor(val, 1, 32);  // column d^1 lives in lane^1
            if (part < 2) {
                float ang = (float)t * invf;
                float sn, cs;
                __sincosf(ang, &sn, &cs);
                float rot = (d & 1) ? partner : -partner;  // rotate_half interleaved
                val = val * cs + rot * sn;
            }
            const size_t bh = (size_t)(b * NH + h);
            if (part == 0)      qo[(bh * TT + t) * HD + d] = (bf16)val;
            else if (part == 1) ko[(bh * TT + t) * HD + d] = (bf16)val;
            else                vo[(bh * HD + d) * TT + t] = (bf16)val;  // transposed
        }
    }
}

// ---------------------------------------------------------------- flash attention
// One wave handles a 16-query tile of one (b,h); streams 32 keys/step.
// V is (B,H,D,T): B-fragment for P@V is a contiguous 32B load per lane.
__global__ void flash_attn_kernel(const bf16* __restrict__ Q, const bf16* __restrict__ K,
                                  const bf16* __restrict__ V, bf16* __restrict__ Y) {
    __shared__ __align__(32) bf16 pbuf[4][16 * 32];  // per-wave P bounce buffer
    const int lane = threadIdx.x & 31;
    const int hi = lane >> 4, ln = lane & 15;
    const int wslot = threadIdx.x >> 5;
    const int wid = blockIdx.x * 4 + wslot;
    const int bh = wid >> 7, qt = wid & 127;
    const int b = bh >> 4, h = bh & 15;
    const int q0 = qt << 4;

    const bf16* Qh = Q + (size_t)bh * TT * HD;
    const bf16* Kh = K + (size_t)bh * TT * HD;
    const bf16* Vh = V + (size_t)bh * HD * TT;       // transposed layout

    // Q A-fragments, 4 chunks of K=32 over HD=128 (held for whole kernel)
    v16bf qf[4];
#pragma unroll
    for (int cc = 0; cc < 4; ++cc)
        qf[cc] = loadA(Qh + (size_t)(q0 + ln) * HD + cc * 32 + hi * 16);

    v8f o[8];
    float m[8], l[8];
#pragma unroll
    for (int nt = 0; nt < 8; ++nt) o[nt] = zero8();
#pragma unroll
    for (int r = 0; r < 8; ++r) { m[r] = -1e30f; l[r] = 0.f; }

    const float scale = 0.08838834764831845f;  // 1/sqrt(128)
    const int jmax = (q0 + 16 + 31) >> 5;      // causal key-step bound

    for (int j = 0; j < jmax; ++j) {
        const int kb = j << 5;
        if (kb + 32 < TT) {
            __builtin_prefetch(Kh + (size_t)(kb + 32 + ln) * HD, 0, 1);
#pragma unroll
            for (int i = 0; i < 4; ++i)        // V rows d = i*32 + lane
                __builtin_prefetch(Vh + (size_t)(i * 32 + lane) * TT + kb + 32, 0, 1);
        }

        // S = Q K^T : two 16x16 subtiles; single base, immediate offsets
        const bf16* kp = Kh + (size_t)(kb + ln) * HD + hi * 16;
        v8f s[2] = {zero8(), zero8()};
#pragma unroll
        for (int sub = 0; sub < 2; ++sub) {
#pragma unroll
            for (int cc = 0; cc < 4; ++cc) {
                v16bf kf = *(const v16bf*)(kp + sub * 16 * HD + cc * 32);
                s[sub] = wmma_bf16(qf[cc], kf, s[sub]);
            }
        }

        // online softmax (row reductions within 16-lane half-groups)
#pragma unroll
        for (int r = 0; r < 8; ++r) {
            const int qrow = q0 + hi * 8 + r;
            const int c0 = kb + ln, c1 = kb + 16 + ln;
            float s0 = (c0 <= qrow) ? s[0][r] * scale : -1e30f;
            float s1 = (c1 <= qrow) ? s[1][r] * scale : -1e30f;
            float mx = fmaxf(s0, s1);
            mx = fmaxf(mx, __shfl_xor(mx, 1, 32));
            mx = fmaxf(mx, __shfl_xor(mx, 2, 32));
            mx = fmaxf(mx, __shfl_xor(mx, 4, 32));
            mx = fmaxf(mx, __shfl_xor(mx, 8, 32));
            const float mnew = fmaxf(m[r], mx);
            const float alpha = __expf(m[r] - mnew);
            const float p0 = __expf(s0 - mnew);
            const float p1 = __expf(s1 - mnew);
            l[r] = l[r] * alpha + p0 + p1;
            m[r] = mnew;
#pragma unroll
            for (int nt = 0; nt < 8; ++nt) o[nt][r] *= alpha;
            // write P (C layout) to LDS row-major [M][k]
            pbuf[wslot][(r + hi * 8) * 32 + ln] = (bf16)p0;
            pbuf[wslot][(r + hi * 8) * 32 + 16 + ln] = (bf16)p1;
        }

        // reload P in A-fragment layout (DS ops are in-order within a wave)
        const bf16* pp = &pbuf[wslot][ln * 32 + hi * 16];
        v16bf pf = cat16(*(const v8bf*)pp, *(const v8bf*)(pp + 8));

        // O += P @ V; V^T rows contiguous in key; depth-1 pipeline over nt
        const bf16* vp = Vh + (size_t)ln * TT + kb + hi * 16;  // nt rows: +nt*16*TT (64KB)
        v16bf vcur = *(const v16bf*)vp;
#pragma unroll
        for (int nt = 0; nt < 8; ++nt) {
            v16bf vnext = vcur;
            if (nt < 7) vnext = *(const v16bf*)(vp + (nt + 1) * 16 * TT);
            o[nt] = wmma_bf16(pf, vcur, o[nt]);
            vcur = vnext;
        }
    }

    // finalize: divide by row sums, write y as (B*T, C) bf16
#pragma unroll
    for (int r = 0; r < 8; ++r) {
        float L = l[r];
        L += __shfl_xor(L, 1, 32);
        L += __shfl_xor(L, 2, 32);
        L += __shfl_xor(L, 4, 32);
        L += __shfl_xor(L, 8, 32);
        const float inv = 1.0f / L;
        const int qrow = q0 + hi * 8 + r;
        const size_t yrow = (size_t)b * TT + qrow;
#pragma unroll
        for (int nt = 0; nt < 8; ++nt)
            Y[yrow * CC + h * HD + nt * 16 + ln] = (bf16)(o[nt][r] * inv);
    }
}

// ---------------------------------------------------------------- output projection
__global__ void gemm_proj_kernel(const bf16* __restrict__ A, const bf16* __restrict__ Bt,
                                 const float* __restrict__ bias, float* __restrict__ out) {
    const int lane = threadIdx.x & 31;
    const int hi = lane >> 4, ln = lane & 15;
    const int wid = blockIdx.x * (blockDim.x >> 5) + (threadIdx.x >> 5);
    const int mt = wid / 32, ng = wid % 32;          // 512 x 32 wave tiles
    const int m0 = mt * 16, n0 = ng * 64;

    v8f acc[4];
#pragma unroll
    for (int nt = 0; nt < 4; ++nt) acc[nt] = zero8();

    const bf16* ap = A + (size_t)(m0 + ln) * CC + hi * 16;
    const bf16* bp = Bt + (size_t)(n0 + ln) * CC + hi * 16;

    v16bf af = loadA(ap);
    v16bf bfr[4];
#pragma unroll
    for (int nt = 0; nt < 4; ++nt) bfr[nt] = *(const v16bf*)(bp + nt * 16 * CC);
    ap += 32;
    bp += 32;

    for (int c = 0; c < CC - 32; c += 32) {
        v16bf af_n = loadA(ap);
        v16bf bfn[4];
#pragma unroll
        for (int nt = 0; nt < 4; ++nt) bfn[nt] = *(const v16bf*)(bp + nt * 16 * CC);
        ap += 32;
        bp += 32;
#pragma unroll
        for (int nt = 0; nt < 4; ++nt) acc[nt] = wmma_bf16(af, bfr[nt], acc[nt]);
        af = af_n;
#pragma unroll
        for (int nt = 0; nt < 4; ++nt) bfr[nt] = bfn[nt];
    }
#pragma unroll
    for (int nt = 0; nt < 4; ++nt) acc[nt] = wmma_bf16(af, bfr[nt], acc[nt]);

#pragma unroll
    for (int nt = 0; nt < 4; ++nt) {
        const int ncol = n0 + nt * 16 + ln;
        const float bv = bias[ncol];
#pragma unroll
        for (int r = 0; r < 8; ++r) {
            const int row = m0 + hi * 8 + r;
            out[(size_t)row * CC + ncol] = acc[nt][r] + bv;
        }
    }
}

// ---------------------------------------------------------------- launcher
extern "C" void kernel_launch(void* const* d_in, const int* in_sizes, int n_in,
                              void* d_out, int out_size, void* d_ws, size_t ws_size,
                              hipStream_t stream) {
    const float* x     = (const float*)d_in[0];
    const float* Wqkv  = (const float*)d_in[1];
    const float* bqkv  = (const float*)d_in[2];
    const float* Wproj = (const float*)d_in[3];
    const float* bproj = (const float*)d_in[4];
    float* out = (float*)d_out;

    bf16* ws = (bf16*)d_ws;
    bf16* xb     = ws;                                  // 8192*2048
    bf16* wqkvT  = xb + (size_t)ROWS * CC;              // 6144*2048
    bf16* wprojT = wqkvT + (size_t)N3C * CC;            // 2048*2048
    bf16* d_q    = wprojT + (size_t)CC * CC;            // 8192*2048 (B,H,T,D)
    bf16* d_k    = d_q + (size_t)ROWS * CC;             // (B,H,T,D)
    bf16* d_v    = d_k + (size_t)ROWS * CC;             // (B,H,D,T) transposed
    bf16* d_y    = d_v + (size_t)ROWS * CC;             // 8192*2048

    cvt_bf16_kernel<<<4096, 256, 0, stream>>>(x, xb, ROWS * CC);
    transpose_bf16_kernel<<<8192, 256, 0, stream>>>(Wqkv, wqkvT, CC, N3C);
    transpose_bf16_kernel<<<4096, 256, 0, stream>>>(Wproj, wprojT, CC, CC);

    // (8192/16)*(6144/64) = 49152 waves, 4 waves/block
    gemm_qkv_rope_kernel<<<12288, 128, 0, stream>>>(xb, wqkvT, bqkv, d_q, d_k, d_v);

    // (4*16)*(2048/16) = 8192 waves, 4 waves/block
    flash_attn_kernel<<<2048, 128, 0, stream>>>(d_q, d_k, d_v, d_y);

    // (8192/16)*(2048/64) = 16384 waves, 4 waves/block
    gemm_proj_kernel<<<4096, 128, 0, stream>>>(d_y, wprojT, bproj, out);
}
